// MambaAttentionBlock_8950711844968
// MI455X (gfx1250) — matrix operand order repaired and checked by reading, not compiled, not run on hardware
//
#include <hip/hip_runtime.h>
#include <hip/hip_bf16.h>

// ---------------------------------------------------------------------------
// Types for CDNA5 WMMA (wave32, v_wmma_f32_16x16x32_bf16)
// ---------------------------------------------------------------------------
typedef __attribute__((ext_vector_type(16))) __bf16 v16bf;
typedef __attribute__((ext_vector_type(8)))  float  v8f;

__device__ __forceinline__ unsigned short f2bf(float f) {
    unsigned int x = __builtin_bit_cast(unsigned int, f);
    unsigned int r = x + 0x7FFFu + ((x >> 16) & 1u);   // RNE
    return (unsigned short)(r >> 16);
}

__device__ __forceinline__ v8f wmma_bf16(v16bf a, v16bf b, v8f c) {
    return __builtin_amdgcn_wmma_f32_16x16x32_bf16(false, a, false, b, (short)0, c, false, false);
}

// ---------------------------------------------------------------------------
// Wide fragment loaders (2 x b128 per fragment).
//
// A 16x32 bf16 fragment (ISA 7.12.2): lane holds row = lane&15 and two
// contiguous 8-elem chunks: k in [kb,kb+8) and [16+kb,16+kb+8), kb = 0|8.
// ---------------------------------------------------------------------------
__device__ __forceinline__ v16bf a_frag_16(const unsigned short* p, size_t strideElems, int lane) {
    v16bf a; uint4* ap = (uint4*)&a;
    int row = lane & 15;
    int kb  = (lane < 16) ? 0 : 8;
    const unsigned short* q = p + (size_t)row * strideElems + kb;
    ap[0] = *(const uint4*)(q);        // k = kb .. kb+7
    ap[1] = *(const uint4*)(q + 16);   // k = 16+kb .. 16+kb+7
    return a;
}

// B 32x16 bf16 fragment: lane holds col = lane&15, two contiguous 8-elem chunks
// k in [kb,kb+8) and [16+kb,16+kb+8), kb = 0|8.  Source: element (k, col) at p[col*stride + k].
__device__ __forceinline__ v16bf b_frag_strided(const unsigned short* p, size_t strideElems, int lane) {
    v16bf b; uint4* bp = (uint4*)&b;
    int col = lane & 15;
    int kb  = (lane < 16) ? 0 : 8;
    const unsigned short* q = p + (size_t)col * strideElems + kb;
    bp[0] = *(const uint4*)(q);
    bp[1] = *(const uint4*)(q + 16);
    return b;
}

// ---------------------------------------------------------------------------
// CDNA5 async global->LDS copy (ASYNCcnt path), 16B per lane, GV addressing.
// ---------------------------------------------------------------------------
__device__ __forceinline__ void async_load_b128(unsigned lds_off, const void* gaddr) {
    asm volatile("global_load_async_to_lds_b128 %0, %1, off"
                 :: "v"(lds_off), "v"(gaddr) : "memory");
}
__device__ __forceinline__ void wait_async0() {
    asm volatile("s_wait_asynccnt 0x0" ::: "memory");
}
__device__ __forceinline__ void wait_async3() {
    asm volatile("s_wait_asynccnt 0x3" ::: "memory");
}
__device__ __forceinline__ void wait_ds0() {
    asm volatile("s_wait_dscnt 0x0" ::: "memory");
}
__device__ __forceinline__ unsigned lds_addr_of(const void* p) {
    return (unsigned)(unsigned long long)p;   // low 32 bits = LDS offset
}

// ---------------------------------------------------------------------------
// Weight convert + transpose: in [K][N] f32 -> out [N][K] bf16 (k-contiguous)
// ---------------------------------------------------------------------------
__global__ void k_cvt_transpose(const float* __restrict__ in, unsigned short* __restrict__ out,
                                int K, int N) {
    int i = blockIdx.x * blockDim.x + threadIdx.x;
    if (i >= K * N) return;
    int n = i / K, k = i % K;
    out[i] = f2bf(in[(size_t)k * N + n]);
}

// ---------------------------------------------------------------------------
// V transpose: vt[b][d][t] = qkv[(b*Nseq+t)*2304 + 1536 + d]   (bf16)
// ---------------------------------------------------------------------------
__global__ void k_vt(const unsigned short* __restrict__ qkv, unsigned short* __restrict__ vt,
                     int Nseq, int total) {
    int i = blockIdx.x * blockDim.x + threadIdx.x;
    if (i >= total) return;
    int t = i % Nseq;
    int d = (i / Nseq) % 768;
    int b = i / (Nseq * 768);
    vt[i] = qkv[((size_t)b * Nseq + t) * 2304 + 1536 + d];
}

// ---------------------------------------------------------------------------
// LayerNorm: one block per token, Dm = 768, emits bf16.
// ---------------------------------------------------------------------------
__global__ __launch_bounds__(256) void k_layernorm(const float* __restrict__ x,
                                                   const float* __restrict__ g,
                                                   const float* __restrict__ b,
                                                   unsigned short* __restrict__ out, int Dm) {
    int t = blockIdx.x;
    const float* xr = x + (size_t)t * Dm;
    __shared__ float red[256];
    float s = 0.f;
    for (int j = threadIdx.x; j < Dm; j += 256) s += xr[j];
    red[threadIdx.x] = s; __syncthreads();
    for (int off = 128; off > 0; off >>= 1) {
        if (threadIdx.x < off) red[threadIdx.x] += red[threadIdx.x + off];
        __syncthreads();
    }
    float mean = red[0] / (float)Dm;
    __syncthreads();
    float v = 0.f;
    for (int j = threadIdx.x; j < Dm; j += 256) { float d = xr[j] - mean; v += d * d; }
    red[threadIdx.x] = v; __syncthreads();
    for (int off = 128; off > 0; off >>= 1) {
        if (threadIdx.x < off) red[threadIdx.x] += red[threadIdx.x + off];
        __syncthreads();
    }
    float inv = rsqrtf(red[0] / (float)Dm + 1e-5f);
    for (int j = threadIdx.x; j < Dm; j += 256)
        out[(size_t)t * Dm + j] = f2bf((xr[j] - mean) * inv * g[j] + b[j]);
}

// ---------------------------------------------------------------------------
// WMMA GEMM, transposed weights, double-buffered async-LDS pipeline:
//   C[M,N] = act(A[M,K](bf16) @ BT[N,K](bf16)^T + bias) (+ res)
// BM=128, BN=64, BK=32; 256 threads = 8 waves; wave computes 32x32
// (4 WMMA / K-step). Tile i+1 streams in via global_load_async_to_lds_b128
// while tile i runs on the matrix units; s_wait_asynccnt 3 retires only the
// previous tile's 3 per-wave async ops (async loads complete in order).
// ACT: 0 = none, 1 = exact GELU.
// ---------------------------------------------------------------------------
template <int ACT>
__global__ __launch_bounds__(256) void k_gemm_bt(const unsigned short* __restrict__ A,
                                                 const unsigned short* __restrict__ BT,
                                                 const float* __restrict__ bias,
                                                 const float* __restrict__ res,
                                                 float* __restrict__ outF,
                                                 unsigned short* __restrict__ outB,
                                                 int M, int N, int K) {
    __shared__ __align__(16) unsigned short lA[2][128][40];   // 2 x 10 KB
    __shared__ __align__(16) unsigned short lB[2][64][40];    // 2 x 5 KB
    int tid = threadIdx.x, lane = tid & 31, w = tid >> 5;
    int wr = w & 3, wc = w >> 2;                 // 4 x 32-row groups, 2 x 32-col groups
    int bm = blockIdx.y * 128, bn = blockIdx.x * 64;
    v8f acc00 = {}, acc01 = {}, acc10 = {}, acc11 = {};

    auto issue = [&](int buf, int k0) {
#pragma unroll
        for (int j = 0; j < 2; ++j) {
            int r = (tid + j * 256) >> 2, c = (tid & 3) * 8;
            async_load_b128(lds_addr_of(&lA[buf][r][c]), A + (size_t)(bm + r) * K + k0 + c);
        }
        int r = tid >> 2, c = (tid & 3) * 8;
        async_load_b128(lds_addr_of(&lB[buf][r][c]), BT + (size_t)(bn + r) * K + k0 + c);
    };

    const int nk = K >> 5;
    issue(0, 0);
    for (int i = 0; i < nk; ++i) {
        if (i + 1 < nk) { issue((i + 1) & 1, (i + 1) << 5); wait_async3(); }
        else            { wait_async0(); }
        __syncthreads();
        int bs = i & 1;
        v16bf a0 = a_frag_16(&lA[bs][wr * 32 + 0][0],  40, lane);
        v16bf a1 = a_frag_16(&lA[bs][wr * 32 + 16][0], 40, lane);
        v16bf b0 = b_frag_strided(&lB[bs][wc * 32 + 0][0],  40, lane);
        v16bf b1 = b_frag_strided(&lB[bs][wc * 32 + 16][0], 40, lane);
        acc00 = wmma_bf16(a0, b0, acc00);
        acc01 = wmma_bf16(a0, b1, acc01);
        acc10 = wmma_bf16(a1, b0, acc10);
        acc11 = wmma_bf16(a1, b1, acc11);
        __syncthreads();   // readers done before buffer is re-filled in i+2
    }

    int half = (lane >= 16) ? 8 : 0;
    int coll = lane & 15;
#pragma unroll
    for (int fr = 0; fr < 2; ++fr) {
#pragma unroll
        for (int fc = 0; fc < 2; ++fc) {
            v8f acc = (fr == 0) ? ((fc == 0) ? acc00 : acc01) : ((fc == 0) ? acc10 : acc11);
#pragma unroll
            for (int i = 0; i < 8; ++i) {
                int row = bm + wr * 32 + fr * 16 + half + i;
                int col = bn + wc * 32 + fc * 16 + coll;
                float v = acc[i];
                if (bias) v += bias[col];
                if (ACT == 1) v = 0.5f * v * (1.0f + erff(v * 0.70710678118f));
                if (res)  v += res[(size_t)row * N + col];
                size_t idx = (size_t)row * N + col;
                if (outF) outF[idx] = v;
                if (outB) outB[idx] = f2bf(v);
            }
        }
    }
}

// ---------------------------------------------------------------------------
// Depthwise causal conv (K=4) + SiLU(conv)*SiLU(z), 4 channels per thread.
// ---------------------------------------------------------------------------
__global__ void k_conv_gate(const float* __restrict__ xz, const float* __restrict__ cw,
                            const float* __restrict__ cb, unsigned short* __restrict__ y,
                            int Nseq, int Bsz) {
    const int NQ = 1536 / 4;
    int idx = blockIdx.x * blockDim.x + threadIdx.x;
    int total = Bsz * Nseq * NQ;
    if (idx >= total) return;
    int cq = idx % NQ;
    int t  = (idx / NQ) % Nseq;
    int b  = idx / (NQ * Nseq);
    int c  = cq * 4;
    size_t rowbase = ((size_t)b * Nseq + t) * 3072;

    float4 wv[4];
#pragma unroll
    for (int j = 0; j < 4; ++j) wv[j] = *(const float4*)(cw + (size_t)(c + j) * 4);
    float ac[4];
    float4 bv = *(const float4*)(cb + c);
    ac[0] = bv.x; ac[1] = bv.y; ac[2] = bv.z; ac[3] = bv.w;

#pragma unroll
    for (int h = 0; h < 4; ++h) {
        int tt = t - 3 + h;
        if (tt >= 0) {
            float4 xv = *(const float4*)(xz + ((size_t)b * Nseq + tt) * 3072 + c);
            const float* xp = (const float*)&xv;
#pragma unroll
            for (int j = 0; j < 4; ++j) ac[j] += xp[j] * ((const float*)&wv[j])[h];
        }
    }
    float4 zv = *(const float4*)(xz + rowbase + 1536 + c);
    const float* zp = (const float*)&zv;
    ushort4 o;
    unsigned short* op = (unsigned short*)&o;
#pragma unroll
    for (int j = 0; j < 4; ++j) {
        float sy = ac[j] / (1.f + __expf(-ac[j]));
        float sz = zp[j] / (1.f + __expf(-zp[j]));
        op[j] = f2bf(sy * sz);
    }
    *(ushort4*)(y + ((size_t)b * Nseq + t) * 1536 + c) = o;
}

// ---------------------------------------------------------------------------
// Flash attention, hd=64, 12 heads, unmasked softmax.
// grid=(Nseq/128, B*12), 256 threads = 8 independent waves (16 q-rows each).
// Q pre-scaled by 1/sqrt(hd); K fragments straight from qkv (dim-contiguous),
// V fragments from pre-transposed vt (token-contiguous); only the P
// C-layout -> A-layout shuffle touches LDS.
// ---------------------------------------------------------------------------
__global__ __launch_bounds__(256) void k_flash_attn(const unsigned short* __restrict__ qkv,
                                                    const unsigned short* __restrict__ vt,
                                                    unsigned short* __restrict__ out,
                                                    int Nseq) {
    __shared__ __align__(16) unsigned short lP[8][16][40];
    int bh = blockIdx.y;
    int b = bh / 12, h = bh % 12;
    int qbase = blockIdx.x * 128;
    int tid = threadIdx.x, lane = tid & 31, w = tid >> 5;
    size_t tokoff = (size_t)b * Nseq;

    // Q fragments, wide from global, pre-scaled by 1/8 (exact power of two)
    const unsigned short* qp = qkv + (tokoff + qbase + w * 16) * 2304 + h * 64;
    v16bf q0 = a_frag_16(qp,      2304, lane);
    v16bf q1 = a_frag_16(qp + 32, 2304, lane);
#pragma unroll
    for (int e = 0; e < 16; ++e) {
        q0[e] = (__bf16)((float)q0[e] * 0.125f);
        q1[e] = (__bf16)((float)q1[e] * 0.125f);
    }

    float m[8], l[8];
#pragma unroll
    for (int i = 0; i < 8; ++i) { m[i] = -1e30f; l[i] = 0.f; }
    v8f o0 = {}, o1 = {}, o2 = {}, o3 = {};

    const unsigned short* vbase0 = vt + ((size_t)b * 768 + h * 64) * Nseq;

    int ntiles = Nseq / 32;
    for (int kt = 0; kt < ntiles; ++kt) {
        const unsigned short* kbase = qkv + (tokoff + (size_t)kt * 32) * 2304 + 768 + h * 64;

        // S(16x32) = (Q/8) @ K^T : 4 WMMA, B-frags straight from global
        v8f s0 = {}, s1 = {};
        s0 = wmma_bf16(q0, b_frag_strided(kbase,                          2304, lane), s0);
        s0 = wmma_bf16(q1, b_frag_strided(kbase + 32,                     2304, lane), s0);
        s1 = wmma_bf16(q0, b_frag_strided(kbase + (size_t)16 * 2304,      2304, lane), s1);
        s1 = wmma_bf16(q1, b_frag_strided(kbase + (size_t)16 * 2304 + 32, 2304, lane), s1);

        // Online softmax; row r = i + (lane>=16 ? 8:0), cols live in the
        // lane's 16-lane half -> xor-shuffle reductions (masks < 16).
#pragma unroll
        for (int i = 0; i < 8; ++i) {
            float a0 = s0[i], a1 = s1[i];
            float tmx = fmaxf(a0, a1);
            for (int msk = 1; msk < 16; msk <<= 1) tmx = fmaxf(tmx, __shfl_xor(tmx, msk, 32));
            float mn = fmaxf(m[i], tmx);
            float sc = __expf(m[i] - mn);
            float p0 = __expf(a0 - mn), p1 = __expf(a1 - mn);
            float rs = p0 + p1;
            for (int msk = 1; msk < 16; msk <<= 1) rs += __shfl_xor(rs, msk, 32);
            l[i] = l[i] * sc + rs;
            m[i] = mn;
            o0[i] *= sc; o1[i] *= sc; o2[i] *= sc; o3[i] *= sc;
            s0[i] = p0; s1[i] = p1;
        }

        // Stage P (C-layout -> row-major) in per-wave LDS; per-wave DS is
        // in-order, wait dscnt before re-reading in A-layout.
        {
            int half = (lane >= 16) ? 8 : 0;
            int col = lane & 15;
#pragma unroll
            for (int i = 0; i < 8; ++i) {
                lP[w][half + i][col]      = f2bf(s0[i]);
                lP[w][half + i][col + 16] = f2bf(s1[i]);
            }
        }
        wait_ds0();
        v16bf pa = a_frag_16(&lP[w][0][0], 40, lane);

        // O += P @ V : 4 WMMA, V B-frags from vt (token-contiguous)
        const unsigned short* vb = vbase0 + kt * 32;
        o0 = wmma_bf16(pa, b_frag_strided(vb + (size_t)0  * Nseq, Nseq, lane), o0);
        o1 = wmma_bf16(pa, b_frag_strided(vb + (size_t)16 * Nseq, Nseq, lane), o1);
        o2 = wmma_bf16(pa, b_frag_strided(vb + (size_t)32 * Nseq, Nseq, lane), o2);
        o3 = wmma_bf16(pa, b_frag_strided(vb + (size_t)48 * Nseq, Nseq, lane), o3);
    }

    {
        int half = (lane >= 16) ? 8 : 0;
        int col = lane & 15;
#pragma unroll
        for (int i = 0; i < 8; ++i) {
            int row = qbase + w * 16 + half + i;
            float invl = 1.f / l[i];
            size_t base = (tokoff + row) * 768 + h * 64;
            out[base + 0  + col] = f2bf(o0[i] * invl);
            out[base + 16 + col] = f2bf(o1[i] * invl);
            out[base + 32 + col] = f2bf(o2[i] * invl);
            out[base + 48 + col] = f2bf(o3[i] * invl);
        }
    }
}

// ---------------------------------------------------------------------------
// Host launch
// ---------------------------------------------------------------------------
extern "C" void kernel_launch(void* const* d_in, const int* in_sizes, int n_in,
                              void* d_out, int out_size, void* d_ws, size_t ws_size,
                              hipStream_t stream) {
    (void)in_sizes; (void)n_in; (void)out_size; (void)ws_size;
    const int Bsz = 2, Nseq = 2048, Dm = 768, Din = 1536, Hid = 3072;
    const int T = Bsz * Nseq;

    const float* x_in  = (const float*)d_in[0];
    const float* ln1_g = (const float*)d_in[1];
    const float* ln1_b = (const float*)d_in[2];
    const float* inpW  = (const float*)d_in[3];
    const float* convW = (const float*)d_in[4];
    const float* convB = (const float*)d_in[5];
    const float* outpW = (const float*)d_in[6];
    const float* ln2_g = (const float*)d_in[7];
    const float* ln2_b = (const float*)d_in[8];
    const float* qkvW  = (const float*)d_in[9];
    const float* qkvB  = (const float*)d_in[10];
    const float* aprW  = (const float*)d_in[11];
    const float* aprB  = (const float*)d_in[12];
    const float* ln3_g = (const float*)d_in[13];
    const float* ln3_b = (const float*)d_in[14];
    const float* f1W   = (const float*)d_in[15];
    const float* f1B   = (const float*)d_in[16];
    const float* f2W   = (const float*)d_in[17];
    const float* f2B   = (const float*)d_in[18];

    char* ws = (char*)d_ws;
    size_t off = 0;
    auto carve = [&](size_t bytes) -> char* {
        char* p = ws + off;
        off = (off + bytes + 255) & ~(size_t)255;
        return p;
    };
    float*          xres  = (float*)         carve((size_t)T * Dm * 4);
    unsigned short* lnbf  = (unsigned short*)carve((size_t)T * Dm * 2);
    float*          xz    = (float*)         carve((size_t)T * 2 * Din * 4); // reused as ffn hidden
    unsigned short* ybf   = (unsigned short*)carve((size_t)T * Din * 2);     // reused as attn out
    unsigned short* qkvbf = (unsigned short*)carve((size_t)T * 3 * Dm * 2);
    unsigned short* vtbf  = (unsigned short*)carve((size_t)Bsz * Dm * Nseq * 2);
    unsigned short* wt_in  = (unsigned short*)carve((size_t)Dm * 2 * Din * 2);
    unsigned short* wt_out = (unsigned short*)carve((size_t)Din * Dm * 2);
    unsigned short* wt_qkv = (unsigned short*)carve((size_t)Dm * 3 * Dm * 2);
    unsigned short* wt_apr = (unsigned short*)carve((size_t)Dm * Dm * 2);
    unsigned short* wt_f1  = (unsigned short*)carve((size_t)Dm * Hid * 2);
    unsigned short* wt_f2  = (unsigned short*)carve((size_t)Hid * Dm * 2);
    unsigned short* hbf    = (unsigned short*)xz;
    unsigned short* attnbf = ybf;

    auto cvtT = [&](const float* src, unsigned short* dst, int K, int N) {
        int n = K * N;
        k_cvt_transpose<<<(n + 255) / 256, 256, 0, stream>>>(src, dst, K, N);
    };
    cvtT(inpW,  wt_in,  Dm, 2 * Din);
    cvtT(outpW, wt_out, Din, Dm);
    cvtT(qkvW,  wt_qkv, Dm, 3 * Dm);
    cvtT(aprW,  wt_apr, Dm, Dm);
    cvtT(f1W,   wt_f1,  Dm, Hid);
    cvtT(f2W,   wt_f2,  Hid, Dm);

    // ---- SSM branch ----
    k_layernorm<<<T, 256, 0, stream>>>(x_in, ln1_g, ln1_b, lnbf, Dm);
    k_gemm_bt<0><<<dim3((2 * Din) / 64, T / 128), 256, 0, stream>>>(
        lnbf, wt_in, nullptr, nullptr, xz, nullptr, T, 2 * Din, Dm);
    {
        int tot = T * (Din / 4);
        k_conv_gate<<<(tot + 255) / 256, 256, 0, stream>>>(xz, convW, convB, ybf, Nseq, Bsz);
    }
    k_gemm_bt<0><<<dim3(Dm / 64, T / 128), 256, 0, stream>>>(
        ybf, wt_out, nullptr, x_in, xres, nullptr, T, Dm, Din);

    // ---- Attention branch ----
    k_layernorm<<<T, 256, 0, stream>>>(xres, ln2_g, ln2_b, lnbf, Dm);
    k_gemm_bt<0><<<dim3((3 * Dm) / 64, T / 128), 256, 0, stream>>>(
        lnbf, wt_qkv, qkvB, nullptr, nullptr, qkvbf, T, 3 * Dm, Dm);
    {
        int tot = Bsz * Dm * Nseq;
        k_vt<<<(tot + 255) / 256, 256, 0, stream>>>(qkvbf, vtbf, Nseq, tot);
    }
    k_flash_attn<<<dim3(Nseq / 128, Bsz * 12), 256, 0, stream>>>(qkvbf, vtbf, attnbf, Nseq);
    k_gemm_bt<0><<<dim3(Dm / 64, T / 128), 256, 0, stream>>>(
        attnbf, wt_apr, aprB, xres, xres, nullptr, T, Dm, Dm);

    // ---- FFN branch ----
    k_layernorm<<<T, 256, 0, stream>>>(xres, ln3_g, ln3_b, lnbf, Dm);
    k_gemm_bt<1><<<dim3(Hid / 64, T / 128), 256, 0, stream>>>(
        lnbf, wt_f1, f1B, nullptr, nullptr, hbf, T, Hid, Dm);
    k_gemm_bt<0><<<dim3(Dm / 64, T / 128), 256, 0, stream>>>(
        hbf, wt_f2, f2B, xres, (float*)d_out, nullptr, T, Dm, Hid);
}
